// TransformerBlock_69784628625969
// MI455X (gfx1250) — matrix operand order repaired and checked by reading, not compiled, not run on hardware
//
#include <hip/hip_runtime.h>
#include <hip/hip_bf16.h>
#include <math.h>

// Problem constants (match reference)
#define BDIM   2
#define SEQ    2048
#define DMODEL 2048
#define NHEAD  16
#define DHEAD  128
#define UPD    5632
#define MTOK   4096      // B*S
#define EPSV   1e-6f

typedef unsigned short ushortT;
typedef __attribute__((ext_vector_type(16))) __bf16  v16bf;
typedef __attribute__((ext_vector_type(8)))  __bf16  bf8v;
typedef __attribute__((ext_vector_type(8)))  float   v8f;
typedef __attribute__((ext_vector_type(4)))  ushortT us4v;

// ---------- bf16 helpers (bit-level RNE) ----------
__device__ __forceinline__ ushortT f32_bf16(float f) {
  union { float f; unsigned u; } c; c.f = f;
  unsigned u = c.u;
  u += 0x7fffu + ((u >> 16) & 1u);
  return (ushortT)(u >> 16);
}
__device__ __forceinline__ bf8v ld_bf8(const ushortT* p) {
  return *reinterpret_cast<const bf8v*>(p);
}
__device__ __forceinline__ v16bf mk16(bf8v lo, bf8v hi) {
  v16bf r;
#pragma unroll
  for (int i = 0; i < 8; ++i) { r[i] = lo[i]; r[i + 8] = hi[i]; }
  return r;
}
// A-fragment (16x32 bf16, row-major source with 'stride' elements per row).
// Lane l (l&15 = row M). elems 0..7 -> K = (l>>4)*8 + i ; elems 8..15 -> K = 16 + (l>>4)*8 + (i-8)
__device__ __forceinline__ v16bf wfragA(const ushortT* base, int stride, int lane) {
  const int m = lane & 15, g = lane >> 4;
  const ushortT* p = base + (size_t)m * stride + g * 8;
  return mk16(ld_bf8(p), ld_bf8(p + 16));
}

__device__ __forceinline__ v8f wmma_bf16(v16bf a, v16bf b, v8f c) {
  return __builtin_amdgcn_wmma_f32_16x16x32_bf16(false, a, false, b, (short)0, c, false, false);
}

// ---------- CDNA5 async global->LDS copy (ASYNCcnt tracked, bypasses VGPRs) ----------
__device__ __forceinline__ unsigned lds_off32(const void* p) {
  // generic pointer to LDS: low 32 bits are the LDS byte offset (flat aperture rule)
  return (unsigned)(uintptr_t)p;
}
__device__ __forceinline__ void async_ld16(unsigned ldsoff, const ushortT* g) {
  asm volatile("global_load_async_to_lds_b128 %0, %1, off"
               :: "v"(ldsoff), "v"(g) : "memory");
}
__device__ __forceinline__ void wait_async0() {
  asm volatile("s_wait_asynccnt 0x0" ::: "memory");
}

// ---------- f32 -> bf16 conversion (vectorized x4) ----------
__global__ void cvt_bf16_k(const float* __restrict__ src, ushortT* __restrict__ dst, int n4) {
  int i = blockIdx.x * blockDim.x + threadIdx.x;
  if (i >= n4) return;
  float4 v = reinterpret_cast<const float4*>(src)[i];
  us4v o;
  o[0] = f32_bf16(v.x); o[1] = f32_bf16(v.y);
  o[2] = f32_bf16(v.z); o[3] = f32_bf16(v.w);
  reinterpret_cast<us4v*>(dst)[i] = o;
}

// ---------- generic bf16 GEMM, WG tile 128x128, 8 waves (2x4), 64x32 per wave ----------
// Async double-buffered LDS, 2x-unrolled software pipeline with compile-time buffer ids.
// Requires K % 64 == 0 (true for all call sites: 2048, 5632).
constexpr int EPI_F32   = 0;  // C f32 row-major [M,N]
constexpr int EPI_QHEAD = 1;  // C bf16 as [B,H,S,DK]
constexpr int EPI_KT    = 2;  // C bf16 as [B,H,DK,S]  (K transposed for attention)

template <int EPI>
__global__ __launch_bounds__(256)
void gemm_bf16_k(const ushortT* __restrict__ A, const ushortT* __restrict__ Bm,
                 const float* __restrict__ bias, void* __restrict__ Cout,
                 int M, int N, int K) {
  __shared__ ushortT As[2][128][40];   // 128x32 tiles, padded, double buffered
  __shared__ ushortT Bs[2][32][136];   // 32x128 tiles, padded, double buffered
  const int tid = threadIdx.x, lane = tid & 31, wid = tid >> 5;
  const int mw = wid >> 2, nw = wid & 3;
  const int m0 = blockIdx.y * 128, n0 = blockIdx.x * 128;

  v8f acc[4][2];
#pragma unroll
  for (int i = 0; i < 4; ++i)
#pragma unroll
    for (int j = 0; j < 2; ++j)
#pragma unroll
      for (int r = 0; r < 8; ++r) acc[i][j][r] = 0.0f;

  const int ar = tid >> 1, ac = (tid & 1) * 16;   // 256 thr cover 128x32
  const int br = tid >> 3, bc = (tid & 7) * 16;   // 256 thr cover 32x128
  const ushortT* Aload = A + (size_t)(m0 + ar) * K + ac;
  const ushortT* Bload = Bm + (size_t)br * N + n0 + bc;
  const unsigned laA[2] = { lds_off32(&As[0][ar][ac]), lds_off32(&As[1][ar][ac]) };
  const unsigned laB[2] = { lds_off32(&Bs[0][br][bc]), lds_off32(&Bs[1][br][bc]) };

  auto stage = [&](int buf, int k0) {
    const ushortT* ap = Aload + k0;
    const ushortT* bp = Bload + (size_t)k0 * N;
    async_ld16(laA[buf],      ap);
    async_ld16(laA[buf] + 16, ap + 8);
    async_ld16(laB[buf],      bp);
    async_ld16(laB[buf] + 16, bp + 8);
  };
  auto compute = [&](int buf) {
    v16bf af[4], bfr[2];
#pragma unroll
    for (int mt = 0; mt < 4; ++mt)
      af[mt] = wfragA(&As[buf][mw * 64 + mt * 16][0], 40, lane);
#pragma unroll
    for (int nt = 0; nt < 2; ++nt) {
      const ushortT* p = &Bs[buf][(lane >> 4) * 8][nw * 32 + nt * 16 + (lane & 15)];  // lane = N col of B
      bf8v lo, hi;
      for (int i = 0; i < 8; ++i) { lo[i] = __builtin_bit_cast(__bf16, p[i * 136]); hi[i] = __builtin_bit_cast(__bf16, p[(16 + i) * 136]); }
      bfr[nt] = mk16(lo, hi);
    }
#pragma unroll
    for (int mt = 0; mt < 4; ++mt)
#pragma unroll
      for (int nt = 0; nt < 2; ++nt)
        acc[mt][nt] = wmma_bf16(af[mt], bfr[nt], acc[mt][nt]);
  };

  stage(0, 0);                       // prologue
  int k0 = 0;
  for (; k0 + 64 < K; k0 += 64) {    // steady state: unconditional staging
    wait_async0(); __syncthreads();
    stage(1, k0 + 32);
    compute(0);
    wait_async0(); __syncthreads();
    stage(0, k0 + 64);
    compute(1);
  }
  // epilogue: last two tiles (k0 == K-64 here)
  wait_async0(); __syncthreads();
  stage(1, k0 + 32);
  compute(0);
  wait_async0(); __syncthreads();
  compute(1);

  const int lr = lane & 15, lg = lane >> 4;
#pragma unroll
  for (int mt = 0; mt < 4; ++mt)
#pragma unroll
    for (int nt = 0; nt < 2; ++nt)
#pragma unroll
      for (int r = 0; r < 8; ++r) {
        int m = m0 + mw * 64 + mt * 16 + r + 8 * lg;
        int n = n0 + nw * 32 + nt * 16 + lr;
        float v = acc[mt][nt][r] + (bias ? bias[n] : 0.0f);
        if (EPI == EPI_F32) {
          reinterpret_cast<float*>(Cout)[(size_t)m * N + n] = v;
        } else {
          int bb = m / SEQ, s = m - bb * SEQ;
          int hh = n / DHEAD, dk = n - hh * DHEAD;
          ushortT bits = f32_bf16(v);
          if (EPI == EPI_QHEAD)
            reinterpret_cast<ushortT*>(Cout)[((size_t)(bb * NHEAD + hh) * SEQ + s) * DHEAD + dk] = bits;
          else
            reinterpret_cast<ushortT*>(Cout)[((size_t)(bb * NHEAD + hh) * DHEAD + dk) * SEQ + s] = bits;
        }
      }
}

// ---------- fused SwiGLU up-projection: act = silu(h@Wg) * (h@Wu), bf16 out ----------
// WG tile: M 128 x N 64 (per gate/up), 8 waves (4x2), 32x32 per wave, dual accumulators.
// Same async double-buffered 2x-unrolled pipeline (K % 64 == 0: K = 2048).
__global__ __launch_bounds__(256)
void gemm_swiglu_k(const ushortT* __restrict__ A, const ushortT* __restrict__ W,
                   ushortT* __restrict__ act, int M, int K) {
  __shared__ ushortT As[2][128][40];
  __shared__ ushortT Bs[2][2][32][72];   // [buf][gate|up] 32x64 tiles
  const int tid = threadIdx.x, lane = tid & 31, wid = tid >> 5;
  const int mw = wid >> 1, nw = wid & 1;
  const int m0 = blockIdx.y * 128, n0 = blockIdx.x * 64;
  const int N2 = 2 * UPD;

  v8f ag[2][2], au[2][2];
#pragma unroll
  for (int i = 0; i < 2; ++i)
#pragma unroll
    for (int j = 0; j < 2; ++j)
#pragma unroll
      for (int r = 0; r < 8; ++r) { ag[i][j][r] = 0.0f; au[i][j][r] = 0.0f; }

  const int ar = tid >> 1, ac = (tid & 1) * 16;
  const int gg = tid >> 7, rm = tid & 127;
  const int br = rm >> 2, bc = (rm & 3) * 16;
  const ushortT* Aload = A + (size_t)(m0 + ar) * K + ac;
  const ushortT* Bload = W + (size_t)br * N2 + n0 + bc + (size_t)gg * UPD;
  const unsigned laA[2] = { lds_off32(&As[0][ar][ac]), lds_off32(&As[1][ar][ac]) };
  const unsigned laB[2] = { lds_off32(&Bs[0][gg][br][bc]), lds_off32(&Bs[1][gg][br][bc]) };

  auto stage = [&](int buf, int k0) {
    const ushortT* ap = Aload + k0;
    const ushortT* bp = Bload + (size_t)k0 * N2;
    async_ld16(laA[buf],      ap);
    async_ld16(laA[buf] + 16, ap + 8);
    async_ld16(laB[buf],      bp);
    async_ld16(laB[buf] + 16, bp + 8);
  };
  auto compute = [&](int buf) {
    v16bf af[2], bg[2], bu[2];
#pragma unroll
    for (int mt = 0; mt < 2; ++mt)
      af[mt] = wfragA(&As[buf][mw * 32 + mt * 16][0], 40, lane);
#pragma unroll
    for (int nt = 0; nt < 2; ++nt) {
      const ushortT* pg = &Bs[buf][0][(lane >> 4) * 8][nw * 32 + nt * 16 + (lane & 15)];
      const ushortT* pu = &Bs[buf][1][(lane >> 4) * 8][nw * 32 + nt * 16 + (lane & 15)];
      bf8v glo, ghi, ulo, uhi;
      for (int i = 0; i < 8; ++i) {
        glo[i] = __builtin_bit_cast(__bf16, pg[i * 72]); ghi[i] = __builtin_bit_cast(__bf16, pg[(16 + i) * 72]);
        ulo[i] = __builtin_bit_cast(__bf16, pu[i * 72]); uhi[i] = __builtin_bit_cast(__bf16, pu[(16 + i) * 72]);
      }
      bg[nt] = mk16(glo, ghi);
      bu[nt] = mk16(ulo, uhi);
    }
#pragma unroll
    for (int mt = 0; mt < 2; ++mt)
#pragma unroll
      for (int nt = 0; nt < 2; ++nt) {
        ag[mt][nt] = wmma_bf16(af[mt], bg[nt], ag[mt][nt]);
        au[mt][nt] = wmma_bf16(af[mt], bu[nt], au[mt][nt]);
      }
  };

  stage(0, 0);
  int k0 = 0;
  for (; k0 + 64 < K; k0 += 64) {
    wait_async0(); __syncthreads();
    stage(1, k0 + 32);
    compute(0);
    wait_async0(); __syncthreads();
    stage(0, k0 + 64);
    compute(1);
  }
  wait_async0(); __syncthreads();
  stage(1, k0 + 32);
  compute(0);
  wait_async0(); __syncthreads();
  compute(1);

  const int lr = lane & 15, lg = lane >> 4;
#pragma unroll
  for (int mt = 0; mt < 2; ++mt)
#pragma unroll
    for (int nt = 0; nt < 2; ++nt)
#pragma unroll
      for (int r = 0; r < 8; ++r) {
        int m = m0 + mw * 32 + mt * 16 + r + 8 * lg;
        int n = n0 + nw * 32 + nt * 16 + lr;
        float g = ag[mt][nt][r], u = au[mt][nt][r];
        float sg = g / (1.0f + __expf(-g));      // silu
        act[(size_t)m * UPD + n] = f32_bf16(sg * u);
      }
}

// ---------- flash attention: one wave = one 16-row Q block of one (b,h) ----------
__global__ __launch_bounds__(256)
void attn_k(const ushortT* __restrict__ Qh, const ushortT* __restrict__ Kt,
            const ushortT* __restrict__ Vh, ushortT* __restrict__ ctxb) {
  __shared__ ushortT Pl[8][16][40];   // per-wave P tile (16 q x 32 kv), bf16, padded
  const int tid = threadIdx.x, lane = tid & 31, wid = tid >> 5;
  const int unit = blockIdx.x * 8 + wid;
  const int qb = unit & (SEQ / 16 - 1);
  const int bh = unit >> 7;                 // SEQ/16 == 128
  const int bb = bh >> 4, hh = bh & 15;
  const float sc = 0.08838834764831845f;    // 1/sqrt(128)
  const int lr = lane & 15, lg = lane >> 4;

  const ushortT* Qp  = Qh + ((size_t)bh * SEQ + qb * 16) * DHEAD;
  const ushortT* Ktp = Kt + (size_t)bh * DHEAD * SEQ;
  const ushortT* Vp  = Vh + (size_t)bh * SEQ * DHEAD;

  // Q A-fragments for the 4 K-chunks (DK=128 = 4x32)
  v16bf qf[4];
#pragma unroll
  for (int c = 0; c < 4; ++c) {
    const ushortT* p = Qp + (size_t)lr * DHEAD + c * 32 + lg * 8;
    qf[c] = mk16(ld_bf8(p), ld_bf8(p + 16));
  }

  v8f o[8]; float mrow[8], lrow[8];
#pragma unroll
  for (int nt = 0; nt < 8; ++nt)
#pragma unroll
    for (int r = 0; r < 8; ++r) o[nt][r] = 0.0f;
#pragma unroll
  for (int r = 0; r < 8; ++r) { mrow[r] = -3.0e38f; lrow[r] = 0.0f; }

  for (int j = 0; j < SEQ; j += 32) {
    v8f s0, s1;
#pragma unroll
    for (int r = 0; r < 8; ++r) { s0[r] = 0.0f; s1[r] = 0.0f; }
    // scores: Q(16x128) x K^T(128x32) via 4 chunks; Kt gives contiguous B-fragments
#pragma unroll
    for (int c = 0; c < 4; ++c) {
      const ushortT* kp = Ktp + (size_t)(c * 32 + lg * 8) * SEQ + j + lr;   // lane = kv col
      bf8v l0, h0, l1, h1;
      for (int i = 0; i < 8; ++i) {
        l0[i] = __builtin_bit_cast(__bf16, kp[(size_t)i * SEQ]);      h0[i] = __builtin_bit_cast(__bf16, kp[(size_t)(16 + i) * SEQ]);
        l1[i] = __builtin_bit_cast(__bf16, kp[(size_t)i * SEQ + 16]); h1[i] = __builtin_bit_cast(__bf16, kp[(size_t)(16 + i) * SEQ + 16]);
      }
      v16bf b0 = mk16(l0, h0);
      v16bf b1 = mk16(l1, h1);
      s0 = wmma_bf16(qf[c], b0, s0);
      s1 = wmma_bf16(qf[c], b1, s1);
    }
    // online softmax (row = fixed M: reduce across the 16-lane N group)
#pragma unroll
    for (int r = 0; r < 8; ++r) {
      float a = s0[r] * sc, b2 = s1[r] * sc;
      float mx = fmaxf(a, b2);
#pragma unroll
      for (int d = 1; d < 16; d <<= 1) mx = fmaxf(mx, __shfl_xor(mx, d, 32));
      float mnew  = fmaxf(mrow[r], mx);
      float alpha = __expf(mrow[r] - mnew);
      float p0 = __expf(a - mnew), p1 = __expf(b2 - mnew);
      float rs = p0 + p1;
#pragma unroll
      for (int d = 1; d < 16; d <<= 1) rs += __shfl_xor(rs, d, 32);
      mrow[r] = mnew;
      lrow[r] = lrow[r] * alpha + rs;
#pragma unroll
      for (int nt = 0; nt < 8; ++nt) o[nt][r] *= alpha;
      Pl[wid][r + 8 * lg][lr]      = f32_bf16(p0);
      Pl[wid][r + 8 * lg][16 + lr] = f32_bf16(p1);
    }
    asm volatile("s_wait_dscnt 0x0" ::: "memory");  // wave-private LDS relayout barrier
    v16bf pf;
    {
      const ushortT* p = &Pl[wid][lr][lg * 8];
      pf = mk16(ld_bf8(p), ld_bf8(p + 16));
    }
    // ctx += P(16x32) x V(32x128)
#pragma unroll
    for (int nt = 0; nt < 8; ++nt) {
      const ushortT* vp = Vp + (size_t)(j + lg * 8) * DHEAD + nt * 16 + lr;  // lane = dk col
      bf8v vl, vh;
      for (int i = 0; i < 8; ++i) { vl[i] = __builtin_bit_cast(__bf16, vp[i * DHEAD]); vh[i] = __builtin_bit_cast(__bf16, vp[(16 + i) * DHEAD]); }
      v16bf bv = mk16(vl, vh);
      o[nt] = wmma_bf16(pf, bv, o[nt]);
    }
  }

  // normalize and store ctx as bf16 [B,S,D] (token-major, ready for @Wo)
#pragma unroll
  for (int r = 0; r < 8; ++r) {
    float inv = 1.0f / lrow[r];
    int s = qb * 16 + r + 8 * lg;
    size_t base = ((size_t)bb * SEQ + s) * DMODEL + (size_t)hh * DHEAD;
#pragma unroll
    for (int nt = 0; nt < 8; ++nt)
      ctxb[base + nt * 16 + lr] = f32_bf16(o[nt][r] * inv);
  }
}

// ---------- residual add + rmsnorm: out_f32 (and optional bf16 copy) ----------
__global__ __launch_bounds__(256)
void add_rmsnorm_k(const float* __restrict__ xa, const float* __restrict__ yb,
                   float* __restrict__ hout, ushortT* __restrict__ hb16) {
  const int row = blockIdx.x;
  const float* pa = xa + (size_t)row * DMODEL;
  const float* pb = yb + (size_t)row * DMODEL;
  float vcache[8];
  float ss = 0.0f;
#pragma unroll
  for (int t = 0; t < 8; ++t) {
    int c = threadIdx.x + t * 256;
    float v = pa[c] + pb[c];
    vcache[t] = v; ss += v * v;
  }
#pragma unroll
  for (int d = 1; d < 32; d <<= 1) ss += __shfl_xor(ss, d, 32);
  __shared__ float red[8];
  if ((threadIdx.x & 31) == 0) red[threadIdx.x >> 5] = ss;
  __syncthreads();
  float tot = 0.0f;
#pragma unroll
  for (int i = 0; i < 8; ++i) tot += red[i];
  float scale = rsqrtf(tot * (1.0f / DMODEL) + EPSV);
#pragma unroll
  for (int t = 0; t < 8; ++t) {
    int c = threadIdx.x + t * 256;
    float v = vcache[t] * scale;
    hout[(size_t)row * DMODEL + c] = v;
    if (hb16) hb16[(size_t)row * DMODEL + c] = f32_bf16(v);
  }
}

// ---------- driver ----------
extern "C" void kernel_launch(void* const* d_in, const int* in_sizes, int n_in,
                              void* d_out, int out_size, void* d_ws, size_t ws_size,
                              hipStream_t stream) {
  const float* x     = (const float*)d_in[0];
  const float* Wq    = (const float*)d_in[1];
  const float* bq    = (const float*)d_in[2];
  const float* Wk    = (const float*)d_in[3];
  const float* bk    = (const float*)d_in[4];
  const float* Wv    = (const float*)d_in[5];
  const float* bv    = (const float*)d_in[6];
  const float* Wo    = (const float*)d_in[7];
  const float* bo    = (const float*)d_in[8];
  const float* Wup   = (const float*)d_in[9];
  const float* Wdown = (const float*)d_in[10];

  char* ws = (char*)d_ws;
  size_t off = 0;
  auto take = [&](size_t bytes) -> char* {
    char* p = ws + off; off += (bytes + 255) & ~(size_t)255; return p;
  };
  const size_t MD = (size_t)MTOK * DMODEL;         // 8,388,608 elems
  const size_t DD = (size_t)DMODEL * DMODEL;
  ushortT* xb   = (ushortT*)take(MD * 2);
  ushortT* Wqb  = (ushortT*)take(DD * 2);
  ushortT* Wkb  = (ushortT*)take(DD * 2);
  ushortT* Wvb  = (ushortT*)take(DD * 2);
  ushortT* Wob  = (ushortT*)take(DD * 2);
  ushortT* Wupb = (ushortT*)take((size_t)DMODEL * 2 * UPD * 2);
  ushortT* Wdb  = (ushortT*)take((size_t)UPD * DMODEL * 2);
  ushortT* Qhb  = (ushortT*)take(MD * 2);
  ushortT* Ktb  = (ushortT*)take(MD * 2);
  ushortT* Vhb  = (ushortT*)take(MD * 2);
  ushortT* ctxb = (ushortT*)take(MD * 2);
  float*   ao   = (float*)take(MD * 4);
  float*   hbuf = (float*)take(MD * 4);
  ushortT* hb16 = (ushortT*)take(MD * 2);
  ushortT* act  = Qhb;   // alias: Q/K/V dead once ctx@Wo done; act is 46.1MB < 50.3MB span
  float*   ffn  = ao;    // alias: attn_out consumed by first rmsnorm

  dim3 blk(256);
  auto cvt = [&](const float* s, ushortT* d, size_t n) {
    int n4 = (int)(n / 4);
    cvt_bf16_k<<<dim3((n4 + 255) / 256), blk, 0, stream>>>(s, d, n4);
  };
  cvt(x, xb, MD);
  cvt(Wq, Wqb, DD);
  cvt(Wk, Wkb, DD);
  cvt(Wv, Wvb, DD);
  cvt(Wo, Wob, DD);
  cvt(Wup, Wupb, (size_t)DMODEL * 2 * UPD);
  cvt(Wdown, Wdb, (size_t)UPD * DMODEL);

  dim3 gqkv(DMODEL / 128, MTOK / 128);   // (16, 32)
  gemm_bf16_k<EPI_QHEAD><<<gqkv, blk, 0, stream>>>(xb, Wqb, bq, Qhb, MTOK, DMODEL, DMODEL);
  gemm_bf16_k<EPI_KT  ><<<gqkv, blk, 0, stream>>>(xb, Wkb, bk, Ktb, MTOK, DMODEL, DMODEL);
  gemm_bf16_k<EPI_QHEAD><<<gqkv, blk, 0, stream>>>(xb, Wvb, bv, Vhb, MTOK, DMODEL, DMODEL);

  attn_k<<<dim3((BDIM * NHEAD * (SEQ / 16)) / 8), blk, 0, stream>>>(Qhb, Ktb, Vhb, ctxb);

  gemm_bf16_k<EPI_F32><<<gqkv, blk, 0, stream>>>(ctxb, Wob, bo, ao, MTOK, DMODEL, DMODEL);

  add_rmsnorm_k<<<dim3(MTOK), blk, 0, stream>>>(x, ao, hbuf, hb16);

  gemm_swiglu_k<<<dim3(UPD / 64, MTOK / 128), blk, 0, stream>>>(hb16, Wupb, act, MTOK, DMODEL);

  gemm_bf16_k<EPI_F32><<<dim3(DMODEL / 128, MTOK / 128), blk, 0, stream>>>(
      act, Wdb, nullptr, ffn, MTOK, DMODEL, UPD);

  add_rmsnorm_k<<<dim3(MTOK), blk, 0, stream>>>(hbuf, ffn, (float*)d_out, nullptr);
}
